// KGPDPLAM_alpha_transformer_1906965480086
// MI455X (gfx1250) — matrix operand order repaired and checked by reading, not compiled, not run on hardware
//
#include <hip/hip_runtime.h>
#include <hip/hip_bf16.h>
#include <math.h>

// ---------------- problem constants (match reference) ----------------
#define BB 4
#define LL 1024
#define DD 512
#define HH 8
#define DKK 64
#define NLAYER 4
#define HD (HH * DKK)   // 512
#define FFD (4 * DD)    // 2048
#define NEGF (-4294967295.0f)

typedef __attribute__((ext_vector_type(16))) _Float16 v16h;
typedef __attribute__((ext_vector_type(8)))  float    v8f;

#if __has_builtin(__builtin_amdgcn_global_load_async_to_lds_b128) && \
    __has_builtin(__builtin_amdgcn_s_wait_asynccnt)
#define HAVE_ASYNC_LDS 1
typedef int v4i_async __attribute__((vector_size(16)));
typedef __attribute__((address_space(1))) v4i_async* glb_v4i;
typedef __attribute__((address_space(3))) v4i_async* lds_v4i;
#else
#define HAVE_ASYNC_LDS 0
#endif

// ---------------- weight conversion: f32 row-major -> f16, WMMA-fragment-swizzled ------
// Tiles of 32(K) x 128(N). Tile (kt, nb) stored contiguously at base = (kt*(N/128)+nb)*4096.
// Within a tile: element (k_local = half*16+kk, n_local = nt*16+nn) at
//   off = ((nt*32 + half*16 + nn) * 16 + kk)
// so a lane's 16-element B fragment is 32 contiguous bytes.
__global__ __launch_bounds__(256) void cvt_w_swizzle(const float* __restrict__ src,
                                                     _Float16* __restrict__ dst,
                                                     int K, int N) {
  size_t total = (size_t)K * N;
  for (size_t e = (size_t)blockIdx.x * 256 + threadIdx.x; e < total;
       e += (size_t)gridDim.x * 256) {
    int k = (int)(e / N), n = (int)(e % N);
    int kt = k >> 5, kl = k & 31, nb = n >> 7, nl = n & 127;
    int half = kl >> 4, kk = kl & 15, nt = nl >> 4, nn = nl & 15;
    size_t base = ((size_t)kt * (N >> 7) + nb) * 4096;
    dst[base + (((nt * 32 + half * 16 + nn) << 4) + kk)] = (_Float16)src[e];
  }
}

// ---------------- GEMM: C[M,N] = A(f32)[M,K] x Bsw(f16, swizzled) + bias, opt GELU -----
// grid = (N/128, M/64), block = 256 (8 waves). Wave w: rows (w>>1)*16, cols (w&1)*64.
// Each wave: 1 A fragment reused over 4 B fragments -> 4 wmma per K-step.
__global__ __launch_bounds__(256) void gemm_f32f16(const float* __restrict__ A,
                                                   const _Float16* __restrict__ Bsw,
                                                   const float* __restrict__ bias,
                                                   float* __restrict__ C,
                                                   int M, int N, int K, int gelu) {
  __shared__ alignas(16) _Float16 sA[64 * 32];
  __shared__ alignas(16) _Float16 sB[4096];
  const int tid = threadIdx.x;
  const int wave = tid >> 5, lane = tid & 31;
  const int mrow = lane & 15, half = lane >> 4;
  const int wr = wave >> 1, wc = wave & 1;
  const int m0 = blockIdx.y * 64;
  const int nblk = blockIdx.x;
  const int ntilesN = N >> 7;

  v8f acc[4] = {};

  const int nKt = K >> 5;
  for (int kt = 0; kt < nKt; ++kt) {
    // ---- stage A: 64x32 f32 -> f16, 8 elems/thread, row-major stride 32 ----
    {
      int e = tid * 8;
      int r = e >> 5, c = e & 31;
      const float4* ap = (const float4*)(A + (size_t)(m0 + r) * K + kt * 32 + c);
      float4 p0 = ap[0], p1 = ap[1];
      _Float16* d = &sA[e];
      d[0] = (_Float16)p0.x; d[1] = (_Float16)p0.y;
      d[2] = (_Float16)p0.z; d[3] = (_Float16)p0.w;
      d[4] = (_Float16)p1.x; d[5] = (_Float16)p1.y;
      d[6] = (_Float16)p1.z; d[7] = (_Float16)p1.w;
    }
    // ---- stage B: contiguous pre-swizzled 8KB tile, 32B per thread ----
    {
      const _Float16* bp = Bsw + ((size_t)kt * ntilesN + nblk) * 4096 + tid * 16;
#if HAVE_ASYNC_LDS
      glb_v4i gsrc = (glb_v4i)bp;
      lds_v4i ldst = (lds_v4i)&sB[tid * 16];
      __builtin_amdgcn_global_load_async_to_lds_b128(gsrc, ldst, 0, 0);
      __builtin_amdgcn_global_load_async_to_lds_b128(gsrc, ldst, 16, 0);
#else
      const int4* s = (const int4*)bp;
      int4* d = (int4*)&sB[tid * 16];
      d[0] = s[0]; d[1] = s[1];
#endif
      if (kt + 1 < nKt) __builtin_prefetch(bp + (size_t)ntilesN * 4096, 0, 1);
    }
#if HAVE_ASYNC_LDS
    __builtin_amdgcn_s_wait_asynccnt(0);
#endif
    __syncthreads();

    v16h a;
    const _Float16* arow = &sA[(wr * 16 + mrow) * 32];
#pragma unroll
    for (int i = 0; i < 8; ++i) a[i]     = arow[16 * half + i];
#pragma unroll
    for (int i = 0; i < 8; ++i) a[8 + i] = arow[16 * half + 8 + i];

#pragma unroll
    for (int nt = 0; nt < 4; ++nt) {
      v16h b;
      const _Float16* fp = &sB[(((wc * 4 + nt) * 32 + half * 16 + mrow) << 4)];
#pragma unroll
      for (int i = 0; i < 16; ++i) b[i] = fp[i];
      acc[nt] = __builtin_amdgcn_wmma_f32_16x16x32_f16(false, a, false, b, (short)0,
                                                       acc[nt], false, false);
    }
    __syncthreads();
  }

#pragma unroll
  for (int nt = 0; nt < 4; ++nt) {
    int nn = nblk * 128 + (wc * 4 + nt) * 16 + mrow;
    float bs = bias[nn];
#pragma unroll
    for (int r = 0; r < 8; ++r) {
      int mm = m0 + wr * 16 + r + 8 * half;
      float v = acc[nt][r] + bs;
      if (gelu) v = 0.5f * v * (1.0f + erff(v * 0.70710678118654752f));
      C[(size_t)mm * N + nn] = v;
    }
  }
}

// ---------------- attention scores: S = (Q Kt)*scale*kb + bb (+pre) (+mask) -------------
// grid = (L/128 key tiles, L/16 query tiles, B*H), block = 256.
// All waves share q-tile -> Q staged once in LDS (f16). K fragment = 4x float4 per lane.
__global__ __launch_bounds__(256) void attn_scores(const float* __restrict__ Q,
                                                   const float* __restrict__ Kb,
                                                   const int* __restrict__ maskPAD,
                                                   const float* __restrict__ embK,
                                                   const float* __restrict__ embB,
                                                   float* __restrict__ scores,
                                                   int hasPre) {
  __shared__ alignas(16) _Float16 sQ[16 * 64];
  const int tid = threadIdx.x;
  const int wave = tid >> 5, lane = tid & 31;
  const int mrow = lane & 15, half = lane >> 4;
  const int n0 = (blockIdx.x * 8 + wave) * 16;   // key tile
  const int q0 = blockIdx.y * 16;                // query tile
  const int bh = blockIdx.z;
  const int b = bh / HH, h = bh % HH;

  // stage Q tile 16x64 (f32 -> f16), 4 elems/thread
  {
    int e = tid * 4;
    int r = e >> 6, c = e & 63;
    float4 p = *(const float4*)(Q + ((size_t)(b * LL) + q0 + r) * HD + h * DKK + c);
    _Float16* d = &sQ[e];
    d[0] = (_Float16)p.x; d[1] = (_Float16)p.y;
    d[2] = (_Float16)p.z; d[3] = (_Float16)p.w;
  }
  __syncthreads();

  const float* kbase = Kb + ((size_t)(b * LL) + n0 + mrow) * HD + h * DKK;

  v8f acc = {};
#pragma unroll
  for (int dk0 = 0; dk0 < DKK; dk0 += 32) {
    v16h a, bf;
    const _Float16* arow = &sQ[mrow * 64 + dk0];
#pragma unroll
    for (int i = 0; i < 8; ++i) a[i]     = arow[16 * half + i];
#pragma unroll
    for (int i = 0; i < 8; ++i) a[8 + i] = arow[16 * half + 8 + i];

    const float4* kp = (const float4*)(kbase + dk0 + 16 * half);
    float4 p0 = kp[0], p1 = kp[1], p2 = kp[2], p3 = kp[3];
    bf[0]=(_Float16)p0.x; bf[1]=(_Float16)p0.y; bf[2]=(_Float16)p0.z; bf[3]=(_Float16)p0.w;
    bf[4]=(_Float16)p1.x; bf[5]=(_Float16)p1.y; bf[6]=(_Float16)p1.z; bf[7]=(_Float16)p1.w;
    bf[8]=(_Float16)p2.x; bf[9]=(_Float16)p2.y; bf[10]=(_Float16)p2.z; bf[11]=(_Float16)p2.w;
    bf[12]=(_Float16)p3.x; bf[13]=(_Float16)p3.y; bf[14]=(_Float16)p3.z; bf[15]=(_Float16)p3.w;

    acc = __builtin_amdgcn_wmma_f32_16x16x32_f16(false, a, false, bf, (short)0, acc,
                                                 false, false);
  }

  const int kcol = n0 + mrow;
#pragma unroll
  for (int r = 0; r < 8; ++r) {
    int qq = q0 + r + 8 * half;
    float d = fabsf((float)(qq - kcol));
    if (d > 7.0f) d = 7.0f + __log2f(fmaxf(d - 7.0f, 1.0f));
    int idx = (int)fminf(d, 14.0f);
    float kb = embK[(size_t)idx * HH + h];
    float bb = embB[(size_t)idx * HH + h];
    size_t off = ((size_t)bh * LL + qq) * LL + kcol;
    float s = acc[r] * 0.125f * kb + bb;           // scale = 1/sqrt(64)
    if (hasPre) s += scores[off];
    if (maskPAD[((size_t)b * LL + qq) * LL + kcol] == 0) s = NEGF;
    scores[off] = s;
  }
}

// ---------------- row softmax over L, f16 output for next WMMA ----------------
__global__ __launch_bounds__(256) void softmax_rows(const float* __restrict__ scores,
                                                    _Float16* __restrict__ alpha) {
  __shared__ float red[256];
  const size_t row = (size_t)blockIdx.x * LL;
  const int t = threadIdx.x;

  float v[4];
  float mx = -3.4e38f;
#pragma unroll
  for (int i = 0; i < 4; ++i) {
    v[i] = scores[row + t + 256 * i];
    mx = fmaxf(mx, v[i]);
  }
  red[t] = mx; __syncthreads();
  for (int s = 128; s > 0; s >>= 1) {
    if (t < s) red[t] = fmaxf(red[t], red[t + s]);
    __syncthreads();
  }
  mx = red[0]; __syncthreads();

  float sum = 0.0f;
#pragma unroll
  for (int i = 0; i < 4; ++i) { v[i] = __expf(v[i] - mx); sum += v[i]; }
  red[t] = sum; __syncthreads();
  for (int s = 128; s > 0; s >>= 1) {
    if (t < s) red[t] += red[t + s];
    __syncthreads();
  }
  float inv = 1.0f / red[0];
#pragma unroll
  for (int i = 0; i < 4; ++i) alpha[row + t + 256 * i] = (_Float16)(v[i] * inv);
}

// ---------------- z = alpha x V ----------------
// grid = (L/32, B*H), block = 256 (8 waves: 2 q-tiles x 4 d-tiles).
// V tile (32x64) staged per K-step in fragment-swizzled LDS layout.
__global__ __launch_bounds__(256) void attn_av(const _Float16* __restrict__ alpha,
                                               const float* __restrict__ V,
                                               float* __restrict__ Z) {
  __shared__ alignas(16) _Float16 sV[4 * 32 * 16]; // 2048
  const int tid = threadIdx.x;
  const int wave = tid >> 5, lane = tid & 31;
  const int mrow = lane & 15, half = lane >> 4;
  const int q0 = blockIdx.x * 32;
  const int bh = blockIdx.y;
  const int b = bh / HH, h = bh % HH;
  const int qt = wave >> 2, dt = wave & 3;

  const _Float16* abase = alpha + ((size_t)bh * LL + q0 + qt * 16 + mrow) * LL;

  v8f acc = {};
  for (int k0 = 0; k0 < LL; k0 += 32) {
    // stage V 32x64 f32 -> swizzled f16; coalesced reads, scattered 16b LDS writes
    {
      int e = tid * 8;
      int kr = e >> 6;          // 0..31 (k row)
      int dc = e & 63;          // start d col (8 consecutive)
      const float4* vp =
          (const float4*)(V + ((size_t)(b * LL) + k0 + kr) * HD + h * DKK + dc);
      float4 p0 = vp[0], p1 = vp[1];
      float vv[8] = {p0.x, p0.y, p0.z, p0.w, p1.x, p1.y, p1.z, p1.w};
      int hh = kr >> 4, kk = kr & 15;
#pragma unroll
      for (int j = 0; j < 8; ++j) {
        int dcol = dc + j;
        int dtj = dcol >> 4, nn = dcol & 15;
        sV[(((dtj * 32 + hh * 16 + nn) << 4) + kk)] = (_Float16)vv[j];
      }
    }
    __syncthreads();

    v16h a, bf;
#pragma unroll
    for (int i = 0; i < 8; ++i) a[i]     = abase[k0 + 16 * half + i];
#pragma unroll
    for (int i = 0; i < 8; ++i) a[8 + i] = abase[k0 + 16 * half + 8 + i];
    const _Float16* fp = &sV[((dt * 32 + half * 16 + mrow) << 4)];
#pragma unroll
    for (int i = 0; i < 16; ++i) bf[i] = fp[i];

    acc = __builtin_amdgcn_wmma_f32_16x16x32_f16(false, a, false, bf, (short)0, acc,
                                                 false, false);
    __syncthreads();
  }
#pragma unroll
  for (int r = 0; r < 8; ++r)
    Z[((size_t)(b * LL) + q0 + qt * 16 + r + 8 * half) * HD + h * DKK + dt * 16 + mrow] =
        acc[r];
}

// ---------------- out = LayerNorm(X + Y) ----------------
__global__ __launch_bounds__(256) void add_layernorm(const float* __restrict__ X,
                                                     const float* __restrict__ Y,
                                                     const float* __restrict__ g,
                                                     const float* __restrict__ bta,
                                                     float* __restrict__ out) {
  __shared__ float red[256];
  const size_t row = (size_t)blockIdx.x * DD;
  const int t = threadIdx.x;

  float a0 = X[row + t]       + Y[row + t];
  float a1 = X[row + t + 256] + Y[row + t + 256];

  red[t] = a0 + a1; __syncthreads();
  for (int s = 128; s > 0; s >>= 1) {
    if (t < s) red[t] += red[t + s];
    __syncthreads();
  }
  float mu = red[0] * (1.0f / DD); __syncthreads();

  float d0 = a0 - mu, d1 = a1 - mu;
  red[t] = d0 * d0 + d1 * d1; __syncthreads();
  for (int s = 128; s > 0; s >>= 1) {
    if (t < s) red[t] += red[t + s];
    __syncthreads();
  }
  float rstd = rsqrtf(red[0] * (1.0f / DD) + 1e-5f);

  out[row + t]       = d0 * rstd * g[t]       + bta[t];
  out[row + t + 256] = d1 * rstd * g[t + 256] + bta[t + 256];
}

// ---------------- host orchestration ----------------
extern "C" void kernel_launch(void* const* d_in, const int* in_sizes, int n_in,
                              void* d_out, int out_size, void* d_ws, size_t ws_size,
                              hipStream_t stream) {
  (void)in_sizes; (void)n_in; (void)out_size; (void)ws_size;
  const float* x       = (const float*)d_in[0];
  const int*   maskPAD = (const int*)d_in[1];
  const float* WQ  = (const float*)d_in[2];
  const float* bQ  = (const float*)d_in[3];
  const float* WK  = (const float*)d_in[4];
  const float* bK  = (const float*)d_in[5];
  const float* WV  = (const float*)d_in[6];
  const float* bV  = (const float*)d_in[7];
  const float* WO  = (const float*)d_in[8];
  const float* bO  = (const float*)d_in[9];
  const float* embK = (const float*)d_in[10];
  const float* embB = (const float*)d_in[11];
  const float* ln1g = (const float*)d_in[12];
  const float* ln1b = (const float*)d_in[13];
  const float* ln2g = (const float*)d_in[14];
  const float* ln2b = (const float*)d_in[15];
  const float* Wf1 = (const float*)d_in[16];
  const float* bf1 = (const float*)d_in[17];
  const float* Wf2 = (const float*)d_in[18];
  const float* bf2 = (const float*)d_in[19];
  float* outp = (float*)d_out;

  size_t off = 0;
  auto alloc = [&](size_t bytes) {
    void* p = (char*)d_ws + off;
    off += (bytes + 255) & ~(size_t)255;
    return p;
  };
  const size_t Wsmall = (size_t)NLAYER * DD * HD;
  const size_t Wbig   = (size_t)NLAYER * DD * FFD;
  const size_t MTOK   = (size_t)BB * LL;

  _Float16* wqh  = (_Float16*)alloc(Wsmall * 2);
  _Float16* wkh  = (_Float16*)alloc(Wsmall * 2);
  _Float16* wvh  = (_Float16*)alloc(Wsmall * 2);
  _Float16* woh  = (_Float16*)alloc(Wsmall * 2);
  _Float16* wf1h = (_Float16*)alloc(Wbig * 2);
  _Float16* wf2h = (_Float16*)alloc(Wbig * 2);
  float* qb = (float*)alloc(MTOK * HD * 4);
  float* kb = (float*)alloc(MTOK * HD * 4);
  float* vb = (float*)alloc(MTOK * HD * 4);
  float* zb = (float*)alloc(MTOK * HD * 4);
  float* ob = (float*)alloc(MTOK * DD * 4);
  float* hb = (float*)alloc(MTOK * DD * 4);
  float* fb = (float*)alloc(MTOK * FFD * 4);
  float* xb = (float*)alloc(MTOK * DD * 4);
  float* scores = (float*)alloc((size_t)BB * HH * LL * LL * 4);
  _Float16* alpha = (_Float16*)alloc((size_t)BB * HH * LL * LL * 2);

  // convert + swizzle weights (per layer, per matrix)
  for (int i = 0; i < NLAYER; ++i) {
    size_t so = (size_t)i * DD * HD;
    size_t fo = (size_t)i * DD * FFD;
    cvt_w_swizzle<<<256, 256, 0, stream>>>(WQ  + so, wqh  + so, DD, HD);
    cvt_w_swizzle<<<256, 256, 0, stream>>>(WK  + so, wkh  + so, DD, HD);
    cvt_w_swizzle<<<256, 256, 0, stream>>>(WV  + so, wvh  + so, DD, HD);
    cvt_w_swizzle<<<256, 256, 0, stream>>>(WO  + so, woh  + so, HD, DD);
    cvt_w_swizzle<<<512, 256, 0, stream>>>(Wf1 + fo, wf1h + fo, DD, FFD);
    cvt_w_swizzle<<<512, 256, 0, stream>>>(Wf2 + fo, wf2h + fo, FFD, DD);
  }

  const dim3 blk(256);
  const dim3 gProj(HD / 128, MTOK / 64);   // N=512
  const dim3 gFF1(FFD / 128, MTOK / 64);   // N=2048
  const dim3 gScore(LL / 128, LL / 16, BB * HH);
  const dim3 gAV(LL / 32, BB * HH);

  const float* xcur = x;
  for (int i = 0; i < NLAYER; ++i) {
    const size_t wso = (size_t)i * DD * HD;
    const size_t wfo = (size_t)i * DD * FFD;

    gemm_f32f16<<<gProj, blk, 0, stream>>>(xcur, wqh + wso, bQ + i * HD, qb,
                                           (int)MTOK, HD, DD, 0);
    gemm_f32f16<<<gProj, blk, 0, stream>>>(xcur, wkh + wso, bK + i * HD, kb,
                                           (int)MTOK, HD, DD, 0);
    gemm_f32f16<<<gProj, blk, 0, stream>>>(xcur, wvh + wso, bV + i * HD, vb,
                                           (int)MTOK, HD, DD, 0);

    attn_scores<<<gScore, blk, 0, stream>>>(qb, kb, maskPAD,
                                            embK + (size_t)i * 15 * HH,
                                            embB + (size_t)i * 15 * HH,
                                            scores, i > 0 ? 1 : 0);
    softmax_rows<<<dim3(BB * HH * LL), blk, 0, stream>>>(scores, alpha);
    attn_av<<<gAV, blk, 0, stream>>>(alpha, vb, zb);

    gemm_f32f16<<<gProj, blk, 0, stream>>>(zb, woh + wso, bO + i * DD, ob,
                                           (int)MTOK, DD, HD, 0);
    add_layernorm<<<dim3((unsigned)MTOK), blk, 0, stream>>>(xcur, ob,
                                                            ln1g + i * DD, ln1b + i * DD, hb);
    gemm_f32f16<<<gFF1, blk, 0, stream>>>(hb, wf1h + wfo, bf1 + i * FFD, fb,
                                          (int)MTOK, FFD, DD, 1);
    gemm_f32f16<<<gProj, blk, 0, stream>>>(fb, wf2h + wfo, bf2 + i * DD, ob,
                                           (int)MTOK, DD, FFD, 0);
    float* dst = (i == NLAYER - 1) ? outp : xb;
    add_layernorm<<<dim3((unsigned)MTOK), blk, 0, stream>>>(hb, ob,
                                                            ln2g + i * DD, ln2b + i * DD, dst);
    xcur = xb;
  }
}
